// Attention_64544768524747
// MI455X (gfx1250) — matrix operand order repaired and checked by reading, not compiled, not run on hardware
//
#include <hip/hip_runtime.h>
#include <hip/hip_bf16.h>

#define SEQ   2048
#define HDIM  1024
#define KT    32           // keys per tile
#define NKT   (SEQ / KT)   // 64 key tiles
#define BATCH 4
#define QSTR  1032         // padded row stride (elements) for sQ/sK
#define VSTR  40           // padded row stride for sVt
#define PSTR  40           // padded row stride for sP
#define SSTR  35           // padded row stride (f32) for shared S tile

typedef __attribute__((ext_vector_type(16))) __bf16 v16bf;
typedef __attribute__((ext_vector_type(2)))  __bf16 v2bf;
typedef __attribute__((ext_vector_type(8)))  float  v8f;

// ---- fp32 -> packed bf16x2 ----
#if __has_builtin(__builtin_amdgcn_cvt_pk_bf16_f32)
static __device__ __forceinline__ unsigned int pk2(float lo, float hi) {
  v2bf r = __builtin_amdgcn_cvt_pk_bf16_f32(lo, hi);
  return __builtin_bit_cast(unsigned int, r);
}
#else
// plain casts: clang lowers fptrunc f32->bf16 natively (RNE)
static __device__ __forceinline__ unsigned int pk2(float lo, float hi) {
  v2bf r;
  r[0] = (__bf16)lo;
  r[1] = (__bf16)hi;
  return __builtin_bit_cast(unsigned int, r);
}
#endif
static __device__ __forceinline__ unsigned short f2bf(float f) {
  return (unsigned short)(pk2(f, f) & 0xFFFFu);
}

// build a 16 x bf16 WMMA operand from two 16-byte chunks
static __device__ __forceinline__ v16bf ld16(const unsigned short* p0,
                                             const unsigned short* p1) {
  union { v16bf v; uint4 q[2]; } u;
  u.q[0] = *(const uint4*)p0;
  u.q[1] = *(const uint4*)p1;
  return u.v;
}
static __device__ __forceinline__ v8f wmma_bf16(v16bf a, v16bf b, v8f c) {
  return __builtin_amdgcn_wmma_f32_16x16x32_bf16(false, a, false, b,
                                                 (short)0, c, false, false);
}
static __device__ __forceinline__ v8f vzero() {
  v8f z;
  #pragma unroll
  for (int j = 0; j < 8; j++) z[j] = 0.0f;
  return z;
}

__global__ __launch_bounds__(128, 1)
void fused_attn_proj(const float* __restrict__ Q, const float* __restrict__ K,
                     const float* __restrict__ V, const float* __restrict__ W,
                     const float* __restrict__ bvec, float* __restrict__ out) {
  __shared__ alignas(16) unsigned short sQ[16 * QSTR];      // Q tile, later O tile
  __shared__ alignas(16) unsigned short sK[16 * QSTR];      // half key tile
  __shared__ alignas(16) unsigned short sVt[HDIM * VSTR];   // V^T tile (h-major)
  __shared__ alignas(16) unsigned short sP[4][16 * PSTR];   // per-wave P tile
  __shared__ alignas(16) float          sS[16 * SSTR];      // cross-wave S tile

  const int tid  = threadIdx.x;
  const int wave = tid >> 5;
  const int lane = tid & 31;
  const int g    = lane >> 4;   // 16-lane group within wave
  const int t    = lane & 15;

  const int b     = blockIdx.x >> 7;
  const int qbase = (blockIdx.x & 127) << 4;
  const int hw    = wave * 256;               // this wave's H / O-column window
  const float scale = 0.03125f;               // 1/sqrt(1024)

  // ---- load Q tile (16x1024 f32 -> bf16 LDS, padded rows) ----
  {
    const float4* Qg = (const float4*)(Q + ((size_t)b * SEQ + qbase) * HDIM);
    for (int i = tid; i < 16 * HDIM / 4; i += 128) {
      int row = i >> 8, c4 = i & 255;         // 256 float4 per row
      float4 v4 = Qg[i];
      *(uint2*)&sQ[row * QSTR + c4 * 4] = make_uint2(pk2(v4.x, v4.y), pk2(v4.z, v4.w));
    }
  }

  v8f acc[16];
  #pragma unroll
  for (int c = 0; c < 16; c++) acc[c] = vzero();
  float m_run[8], l_run[8];
  #pragma unroll
  for (int r = 0; r < 8; r++) { m_run[r] = -__builtin_inff(); l_run[r] = 0.0f; }

  const unsigned short* aRow = sQ + t * QSTR;
  const unsigned short* bRow = sK + t * QSTR;

  #pragma unroll 1
  for (int kt = 0; kt < NKT; kt++) {
    // ---- stage keys [kt*KT, +16), V^T tile; zero the shared S tile ----
    __syncthreads();
    for (int i = tid; i < 16 * SSTR; i += 128) sS[i] = 0.0f;
    {
      const float4* Kg = (const float4*)(K + ((size_t)b * SEQ + kt * KT) * HDIM);
      for (int i = tid; i < 16 * HDIM / 4; i += 128) {
        int row = i >> 8, c4 = i & 255;
        float4 v4 = Kg[i];
        *(uint2*)&sK[row * QSTR + c4 * 4] = make_uint2(pk2(v4.x, v4.y), pk2(v4.z, v4.w));
      }
      const float4* Vg = (const float4*)(V + ((size_t)b * SEQ + kt * KT) * HDIM);
      for (int i = tid; i < KT * HDIM / 4; i += 128) {
        int e = i * 4, key = e / HDIM, h = e % HDIM;
        float4 v4 = Vg[i];
        unsigned int p01 = pk2(v4.x, v4.y);
        unsigned int p23 = pk2(v4.z, v4.w);
        sVt[(h + 0) * VSTR + key] = (unsigned short)p01;
        sVt[(h + 1) * VSTR + key] = (unsigned short)(p01 >> 16);
        sVt[(h + 2) * VSTR + key] = (unsigned short)p23;
        sVt[(h + 3) * VSTR + key] = (unsigned short)(p23 >> 16);
      }
    }
    __syncthreads();

    if (kt + 1 < NKT) {                        // global_prefetch_b8 next tile
      const char* nk = (const char*)(K + ((size_t)b * SEQ + (kt + 1) * KT) * HDIM);
      const char* nv = (const char*)(V + ((size_t)b * SEQ + (kt + 1) * KT) * HDIM);
      __builtin_prefetch(nk + tid * 1024, 0, 1);
      __builtin_prefetch(nv + tid * 1024, 0, 1);
    }

    // ---- partial S0 = Q[:,hw:hw+256] . K[0:16,hw:hw+256]^T (this wave) ----
    v8f s0p = vzero();
    #pragma unroll
    for (int kbl = 0; kbl < 256; kbl += 32) {
      int kb = hw + kbl;
      v16bf a  = ld16(aRow + kb + g * 16, aRow + kb + g * 16 + 8);   // A layout
      v16bf bb = ld16(bRow + kb + g * 16, bRow + kb + g * 16 + 8);   // B layout
      s0p = wmma_bf16(a, bb, s0p);
    }

    // ---- stage keys [kt*KT+16, +16), then partial S1 ----
    __syncthreads();
    {
      const float4* Kg = (const float4*)(K + ((size_t)b * SEQ + kt * KT + 16) * HDIM);
      for (int i = tid; i < 16 * HDIM / 4; i += 128) {
        int row = i >> 8, c4 = i & 255;
        float4 v4 = Kg[i];
        *(uint2*)&sK[row * QSTR + c4 * 4] = make_uint2(pk2(v4.x, v4.y), pk2(v4.z, v4.w));
      }
    }
    __syncthreads();

    v8f s1p = vzero();
    #pragma unroll
    for (int kbl = 0; kbl < 256; kbl += 32) {
      int kb = hw + kbl;
      v16bf a  = ld16(aRow + kb + g * 16, aRow + kb + g * 16 + 8);
      v16bf bb = ld16(bRow + kb + g * 16, bRow + kb + g * 16 + 8);
      s1p = wmma_bf16(a, bb, s1p);
    }

    // ---- combine the 4 per-wave partials via LDS float atomics ----
    #pragma unroll
    for (int r = 0; r < 8; r++) {              // row m = r + 8*g, col n = t
      atomicAdd(&sS[(r + 8 * g) * SSTR + t],      s0p[r]);   // ds_add_f32
      atomicAdd(&sS[(r + 8 * g) * SSTR + 16 + t], s1p[r]);
    }
    __syncthreads();

    // ---- online softmax over this tile's 32 keys ----
    float alpha[8];
    unsigned short* pw = sP[wave];
    #pragma unroll
    for (int r = 0; r < 8; r++) {
      float x0 = sS[(r + 8 * g) * SSTR + t]      * scale;
      float x1 = sS[(r + 8 * g) * SSTR + 16 + t] * scale;
      float mx = fmaxf(x0, x1);
      mx = fmaxf(mx, __shfl_xor(mx, 1));
      mx = fmaxf(mx, __shfl_xor(mx, 2));
      mx = fmaxf(mx, __shfl_xor(mx, 4));
      mx = fmaxf(mx, __shfl_xor(mx, 8));       // row max within 16-lane group
      float mn = fmaxf(m_run[r], mx);
      float al = __expf(m_run[r] - mn);
      float p0 = __expf(x0 - mn), p1 = __expf(x1 - mn);
      float ps = p0 + p1;
      ps += __shfl_xor(ps, 1);
      ps += __shfl_xor(ps, 2);
      ps += __shfl_xor(ps, 4);
      ps += __shfl_xor(ps, 8);                 // row sum
      l_run[r] = l_run[r] * al + ps;
      m_run[r] = mn;
      alpha[r] = al;
      unsigned short* pr = pw + (r + 8 * g) * PSTR;   // C-layout -> LDS row-major
      pr[t]      = f2bf(p0);
      pr[t + 16] = f2bf(p1);
    }

    // ---- rescale running O, then O += P . V (wave's 256-wide H window) ----
    v16bf pa = ld16(pw + t * PSTR + g * 16, pw + t * PSTR + g * 16 + 8);  // P as A
    #pragma unroll
    for (int c = 0; c < 16; c++) {
      #pragma unroll
      for (int r = 0; r < 8; r++) acc[c][r] *= alpha[r];
      const unsigned short* vRow = sVt + (hw + c * 16 + t) * VSTR;
      v16bf bv = ld16(vRow + g * 16, vRow + g * 16 + 8);
      acc[c] = wmma_bf16(pa, bv, acc[c]);
    }
  }

  // ---- finalize: O /= l, store bf16 O into sQ (Q tile no longer needed) ----
  __syncthreads();
  float inv[8];
  #pragma unroll
  for (int r = 0; r < 8; r++) inv[r] = 1.0f / l_run[r];
  #pragma unroll
  for (int c = 0; c < 16; c++) {
    #pragma unroll
    for (int r = 0; r < 8; r++)
      sQ[(r + 8 * g) * QSTR + hw + c * 16 + t] = f2bf(acc[c][r] * inv[r]);
  }
  __syncthreads();

  // ---- projection: out = O . W^T + b ; wave computes cols [hw, hw+256) ----
  v8f pacc[16];
  #pragma unroll
  for (int c = 0; c < 16; c++) pacc[c] = vzero();
  #pragma unroll 1
  for (int kb = 0; kb < HDIM; kb += 32) {
    v16bf a = ld16(aRow + kb + g * 16, aRow + kb + g * 16 + 8);   // O chunk as A
    #pragma unroll
    for (int oc = 0; oc < 16; oc++) {
      // B(k=h, n=o): lane n = t -> W row o, 16 consecutive h starting g*16
      const float4* wr = (const float4*)(W + (size_t)(hw + oc * 16 + t) * HDIM
                                           + kb + g * 16);
      union { v16bf v; unsigned int u[8]; } ub;
      #pragma unroll
      for (int j = 0; j < 4; j++) {
        float4 wv = wr[j];
        ub.u[2 * j]     = pk2(wv.x, wv.y);
        ub.u[2 * j + 1] = pk2(wv.z, wv.w);
      }
      pacc[oc] = wmma_bf16(a, ub.v, pacc[oc]);
    }
  }

  #pragma unroll
  for (int oc = 0; oc < 16; oc++) {
    int o = hw + oc * 16 + t;
    float bo = bvec[o];
    #pragma unroll
    for (int r = 0; r < 8; r++)
      out[((size_t)b * SEQ + qbase + r + 8 * g) * HDIM + o] = pacc[oc][r] + bo;
  }
}

extern "C" void kernel_launch(void* const* d_in, const int* in_sizes, int n_in,
                              void* d_out, int out_size, void* d_ws, size_t ws_size,
                              hipStream_t stream) {
  const float* q    = (const float*)d_in[0];
  const float* k    = (const float*)d_in[1];
  const float* v    = (const float*)d_in[2];
  const float* w    = (const float*)d_in[3];
  const float* bias = (const float*)d_in[4];
  float* out = (float*)d_out;
  (void)in_sizes; (void)n_in; (void)out_size; (void)d_ws; (void)ws_size;

  dim3 grid(BATCH * (SEQ / 16));   // 512 workgroups
  dim3 block(128);                 // 4 wave32s
  fused_attn_proj<<<grid, block, 0, stream>>>(q, k, v, w, bias, out);
}
